// GATConv_36627481101078
// MI455X (gfx1250) — compile-verified, matrix operand order checked by reading, not run on hardware
//
#include <hip/hip_runtime.h>
#include <hip/hip_bf16.h>

// ---------------- problem constants (from reference) ----------------
constexpr int N_NODES = 100000;
constexpr int N_EDGES = 1600000;
constexpr int IN_F    = 128;
constexpr int HID     = 32;
constexpr int NCLS    = 8;
constexpr int HEADS   = 2;
#define NEG_SLOPE 0.2f

// Layer-1: M = N_NODES rows, K=128, 64 cols (2 heads * 32)
// Layer-2: M = 2*N_NODES rows, K=32,  16 cols (2 heads * 8)
constexpr int M1_TILES = N_NODES / 16;        // 6250 (exact)
constexpr int M2_TILES = (N_NODES * 2) / 16;  // 12500 (exact)

typedef __attribute__((ext_vector_type(16))) __bf16 v16bf;
typedef __attribute__((ext_vector_type(8)))  float  v8f;

union FragU { v16bf v; uint4 q[2]; };

// ---------------- small helpers ----------------
__device__ __forceinline__ unsigned encF(float f) {
    unsigned u = __float_as_uint(f);
    return (u & 0x80000000u) ? ~u : (u | 0x80000000u);   // order-preserving
}
__device__ __forceinline__ float decF(unsigned u) {
    return __uint_as_float((u & 0x80000000u) ? (u ^ 0x80000000u) : ~u);
}
__device__ __forceinline__ float lrelu(float x) {
    return x > 0.0f ? x : NEG_SLOPE * x;
}

// Load a 16x16x32 bf16 A/B operand fragment.
// Row-major source, row length = ldk (bf16 elems).  Per ISA 7.12.2:
//   lanes 0-15 : K = {0..7, 16..23}, lanes 16-31 : K = {8..15, 24..31}
__device__ __forceinline__ v16bf load_frag_bf16(const __bf16* __restrict__ base,
                                                int row, int ldk, int kbase, int lane) {
    const __bf16* p = base + (size_t)row * ldk + kbase + ((lane & 16) ? 8 : 0);
    FragU f;
    f.q[0] = *(const uint4*)(p);
    f.q[1] = *(const uint4*)(p + 16);
    return f.v;
}

// ---------------- generic zero fill ----------------
__global__ void k_zero_f32(float* __restrict__ p, int n) {
    int i = blockIdx.x * blockDim.x + threadIdx.x;
    int stride = gridDim.x * blockDim.x;
    for (; i < n; i += stride) p[i] = 0.0f;
}

// ---------------- pack / convert ----------------
__global__ void k_convert_in(const float* __restrict__ x, __bf16* __restrict__ y, int n) {
    int i = blockIdx.x * blockDim.x + threadIdx.x;
    int stride = gridDim.x * blockDim.x;
    for (; i < n; i += stride) y[i] = (__bf16)x[i];
}

// transpose + convert both weight matrices (tiny, single block)
__global__ void k_pack_weights(const float* __restrict__ W1,  // (128,64) row-major
                               const float* __restrict__ W2,  // (32,16)  row-major
                               __bf16* __restrict__ W1T,      // (64,128)
                               __bf16* __restrict__ W2T) {    // (16,32)
    for (int i = threadIdx.x; i < 64 * 128; i += blockDim.x) {
        int n = i >> 7, k = i & 127;
        W1T[i] = (__bf16)W1[k * 64 + n];
    }
    for (int i = threadIdx.x; i < 16 * 32; i += blockDim.x) {
        int n = i >> 5, k = i & 31;
        W2T[i] = (__bf16)W2[k * 16 + n];
    }
}

// ---------------- WMMA GEMM layer 1: feat1 = A(100000x128) @ W1(128x64) ----------------
__global__ void k_gemm1(const __bf16* __restrict__ A, const __bf16* __restrict__ WT,
                        float* __restrict__ out) {
    int wave = (blockIdx.x * blockDim.x + threadIdx.x) >> 5;
    int lane = threadIdx.x & 31;
    if (wave >= M1_TILES) return;
    int rowA = wave * 16 + (lane & 15);

    v8f acc[4] = {};
#pragma unroll
    for (int kc = 0; kc < 4; ++kc) {
        int kb = kc * 32;
        v16bf av = load_frag_bf16(A, rowA, 128, kb, lane);
#pragma unroll
        for (int ct = 0; ct < 4; ++ct) {
            v16bf bv = load_frag_bf16(WT, ct * 16 + (lane & 15), 128, kb, lane);
            acc[ct] = __builtin_amdgcn_wmma_f32_16x16x32_bf16(
                false, av, false, bv, (short)0, acc[ct], false, false);
        }
    }
    // D layout: VGPR r holds M=r (lanes 0-15), M=r+8 (lanes 16-31); col = lane%16
    int rbase = wave * 16 + ((lane & 16) ? 8 : 0);
    int col = lane & 15;
#pragma unroll
    for (int ct = 0; ct < 4; ++ct)
#pragma unroll
        for (int r = 0; r < 8; ++r)
            out[(size_t)(rbase + r) * 64 + ct * 16 + col] = acc[ct][r];
}

// ---------------- WMMA GEMM layer 2: feat2 = H(200000x32) @ W2(32x16) ----------------
__global__ void k_gemm2(const __bf16* __restrict__ A, const __bf16* __restrict__ WT,
                        float* __restrict__ out) {
    int wave = (blockIdx.x * blockDim.x + threadIdx.x) >> 5;
    int lane = threadIdx.x & 31;
    if (wave >= M2_TILES) return;
    int rowA = wave * 16 + (lane & 15);

    v16bf av = load_frag_bf16(A, rowA, 32, 0, lane);
    v16bf bv = load_frag_bf16(WT, lane & 15, 32, 0, lane);
    v8f acc = {};
    acc = __builtin_amdgcn_wmma_f32_16x16x32_bf16(
        false, av, false, bv, (short)0, acc, false, false);

    int rbase = wave * 16 + ((lane & 16) ? 8 : 0);
    int col = lane & 15;
#pragma unroll
    for (int r = 0; r < 8; ++r)
        out[(size_t)(rbase + r) * 16 + col] = acc[r];
}

// ---------------- per-node attention logits ----------------
__global__ void k_logits1(const float* __restrict__ feat1, const float* __restrict__ al,
                          const float* __restrict__ ar, float* __restrict__ el,
                          float* __restrict__ er) {
    int idx = blockIdx.x * blockDim.x + threadIdx.x;
    if (idx >= N_NODES * HEADS) return;
    int n = idx >> 1, h = idx & 1;
    const float* f = feat1 + (size_t)n * 64 + h * 32;
    float sl = 0.f, sr = 0.f;
#pragma unroll
    for (int j = 0; j < 32; ++j) {
        float v = f[j];
        sl += v * al[h * 32 + j];
        sr += v * ar[h * 32 + j];
    }
    el[idx] = sl;
    er[idx] = sr;
}

__global__ void k_logits2(const float* __restrict__ feat2, const float* __restrict__ al,
                          const float* __restrict__ ar, float* __restrict__ el,
                          float* __restrict__ er) {
    int idx = blockIdx.x * blockDim.x + threadIdx.x;
    if (idx >= N_NODES * 4) return;          // (n, h1, h2)
    int n = idx >> 2, hh = idx & 3, h2 = hh & 1;
    const float* f = feat2 + (size_t)n * 32 + hh * 8;
    float sl = 0.f, sr = 0.f;
#pragma unroll
    for (int c = 0; c < 8; ++c) {
        float v = f[c];
        sl += v * al[h2 * 8 + c];
        sr += v * ar[h2 * 8 + c];
    }
    el[idx] = sl;
    er[idx] = sr;
}

// ---------------- edge passes (C = #channels: 2 for layer1, 4 for layer2) ----------------
template <int C>
__global__ void k_edge_max(const int* __restrict__ src, const int* __restrict__ dst,
                           const float* __restrict__ el, const float* __restrict__ er,
                           unsigned* __restrict__ emax) {
    int i = blockIdx.x * blockDim.x + threadIdx.x;
    if (i >= N_EDGES) return;
    int s = src[i], d = dst[i];
#pragma unroll
    for (int h = 0; h < C; ++h) {
        float e = lrelu(el[(size_t)s * C + h] + er[(size_t)d * C + h]);
        atomicMax(&emax[(size_t)d * C + h], encF(e));
    }
}

template <int C>
__global__ void k_edge_exp(const int* __restrict__ src, const int* __restrict__ dst,
                           const float* __restrict__ el, const float* __restrict__ er,
                           const unsigned* __restrict__ emax, float* __restrict__ asum,
                           float* __restrict__ aout) {
    int i = blockIdx.x * blockDim.x + threadIdx.x;
    if (i >= N_EDGES) return;
    int s = src[i], d = dst[i];
#pragma unroll
    for (int h = 0; h < C; ++h) {
        float e = lrelu(el[(size_t)s * C + h] + er[(size_t)d * C + h]);
        float a = __expf(e - decF(emax[(size_t)d * C + h]));
        unsafeAtomicAdd(&asum[(size_t)d * C + h], a);
        aout[(size_t)i * C + h] = a;
    }
}

// layer-1 aggregation: one wave per edge, lane l handles cols l (head0) and l+32 (head1)
__global__ void k_edge_agg1(const int* __restrict__ src, const int* __restrict__ dst,
                            const float* __restrict__ a, const float* __restrict__ feat1,
                            float* __restrict__ out1) {
    int gid = blockIdx.x * blockDim.x + threadIdx.x;
    int e = gid >> 5, lane = gid & 31;
    if (e >= N_EDGES) return;
    int s = src[e], d = dst[e];
    float a0 = a[(size_t)e * 2];
    float a1 = a[(size_t)e * 2 + 1];
    const float* fs = feat1 + (size_t)s * 64;
    float* od = out1 + (size_t)d * 64;
    unsafeAtomicAdd(&od[lane],      a0 * fs[lane]);
    unsafeAtomicAdd(&od[lane + 32], a1 * fs[lane + 32]);
}

// layer-2 aggregation: one wave per edge, lane l -> channel (l>>3), value l
__global__ void k_edge_agg2(const int* __restrict__ src, const int* __restrict__ dst,
                            const float* __restrict__ a, const float* __restrict__ feat2,
                            float* __restrict__ out2) {
    int gid = blockIdx.x * blockDim.x + threadIdx.x;
    int e = gid >> 5, lane = gid & 31;
    if (e >= N_EDGES) return;
    int s = src[e], d = dst[e];
    float av = a[(size_t)e * 4 + (lane >> 3)];
    unsafeAtomicAdd(&out2[(size_t)d * 32 + lane], av * feat2[(size_t)s * 32 + lane]);
}

// ---------------- finalize ----------------
// h = relu(out1/asum + b1), stored bf16 as rows of length 32 (M = 2N), same linear index
__global__ void k_final1(const float* __restrict__ out1, const float* __restrict__ asum,
                         const float* __restrict__ b1, __bf16* __restrict__ hbf) {
    int idx = blockIdx.x * blockDim.x + threadIdx.x;
    if (idx >= N_NODES * 64) return;
    int n = idx >> 6, r = idx & 63, h = r >> 5, j = r & 31;
    float s = asum[(size_t)n * 2 + h];
    float v = (s > 0.0f) ? out1[idx] / s : 0.0f;
    v += b1[h * 32 + j];
    v = v > 0.0f ? v : 0.0f;
    hbf[idx] = (__bf16)v;
}

// out = out2/asum2 + b2 (in-place on d_out)
__global__ void k_final2(float* __restrict__ out, const float* __restrict__ asum,
                         const float* __restrict__ b2) {
    int idx = blockIdx.x * blockDim.x + threadIdx.x;
    if (idx >= N_NODES * 32) return;
    int n = idx >> 5, r = idx & 31, hh = r >> 3, h2 = hh & 1, c = r & 7;
    float s = asum[(size_t)n * 4 + hh];
    float v = (s > 0.0f) ? out[idx] / s : 0.0f;
    out[idx] = v + b2[h2 * 8 + c];
}

// ---------------- launch ----------------
extern "C" void kernel_launch(void* const* d_in, const int* in_sizes, int n_in,
                              void* d_out, int out_size, void* d_ws, size_t ws_size,
                              hipStream_t stream) {
    const float* in_feat = (const float*)d_in[0];
    const int*   src     = (const int*)d_in[1];
    const int*   dst     = (const int*)d_in[2];
    const float* W1      = (const float*)d_in[3];
    const float* al1     = (const float*)d_in[4];
    const float* ar1     = (const float*)d_in[5];
    const float* b1      = (const float*)d_in[6];
    const float* W2      = (const float*)d_in[7];
    const float* al2     = (const float*)d_in[8];
    const float* ar2     = (const float*)d_in[9];
    const float* b2      = (const float*)d_in[10];
    float* out = (float*)d_out;

    // ---- workspace carve (256B aligned regions) ----
    char* w = (char*)d_ws;
    size_t off = 0;
    auto take = [&](size_t bytes) -> void* {
        void* p = (void*)(w + off);
        off += (bytes + 255) & ~(size_t)255;
        return p;
    };
    __bf16*   in_bf  = (__bf16*)  take((size_t)N_NODES * IN_F * 2);
    __bf16*   w1t    = (__bf16*)  take((size_t)64 * 128 * 2);
    __bf16*   w2t    = (__bf16*)  take((size_t)16 * 32 * 2);
    float*    feat1  = (float*)   take((size_t)N_NODES * 64 * 4);
    float*    el1    = (float*)   take((size_t)N_NODES * 2 * 4);
    float*    er1    = (float*)   take((size_t)N_NODES * 2 * 4);
    unsigned* emax1  = (unsigned*)take((size_t)N_NODES * 2 * 4);
    float*    asum1  = (float*)   take((size_t)N_NODES * 2 * 4);
    float*    a1     = (float*)   take((size_t)N_EDGES * 2 * 4);
    float*    out1   = (float*)   take((size_t)N_NODES * 64 * 4);
    __bf16*   h1bf   = (__bf16*)  take((size_t)N_NODES * 64 * 2);
    float*    feat2  = (float*)   take((size_t)N_NODES * 32 * 4);
    float*    el2    = (float*)   take((size_t)N_NODES * 4 * 4);
    float*    er2    = (float*)   take((size_t)N_NODES * 4 * 4);
    unsigned* emax2  = (unsigned*)take((size_t)N_NODES * 4 * 4);
    float*    asum2  = (float*)   take((size_t)N_NODES * 4 * 4);
    float*    a2     = (float*)   take((size_t)N_EDGES * 4 * 4);
    (void)ws_size; (void)n_in; (void)in_sizes; (void)out_size;

    auto cdiv = [](long a, long b) { return (int)((a + b - 1) / b); };
    const int T = 256;

    // ---- zero accumulators (poisoned between runs) ----
    k_zero_f32<<<cdiv(N_NODES * 64, T), T, 0, stream>>>(out1, N_NODES * 64);
    k_zero_f32<<<cdiv(N_NODES * 2, T), T, 0, stream>>>(asum1, N_NODES * 2);
    k_zero_f32<<<cdiv(N_NODES * 2, T), T, 0, stream>>>((float*)emax1, N_NODES * 2);
    k_zero_f32<<<cdiv(N_NODES * 4, T), T, 0, stream>>>(asum2, N_NODES * 4);
    k_zero_f32<<<cdiv(N_NODES * 4, T), T, 0, stream>>>((float*)emax2, N_NODES * 4);
    k_zero_f32<<<cdiv(N_NODES * 32, T), T, 0, stream>>>(out, N_NODES * 32);

    // ---- convert inputs / pack weights ----
    k_convert_in<<<4096, T, 0, stream>>>(in_feat, in_bf, N_NODES * IN_F);
    k_pack_weights<<<1, T, 0, stream>>>(W1, W2, w1t, w2t);

    // ---- layer 1 ----
    k_gemm1<<<cdiv((long)M1_TILES * 32, T), T, 0, stream>>>(in_bf, w1t, feat1);
    k_logits1<<<cdiv(N_NODES * 2, T), T, 0, stream>>>(feat1, al1, ar1, el1, er1);
    k_edge_max<2><<<cdiv(N_EDGES, T), T, 0, stream>>>(src, dst, el1, er1, emax1);
    k_edge_exp<2><<<cdiv(N_EDGES, T), T, 0, stream>>>(src, dst, el1, er1, emax1, asum1, a1);
    k_edge_agg1<<<cdiv((long)N_EDGES * 32, T), T, 0, stream>>>(src, dst, a1, feat1, out1);
    k_final1<<<cdiv(N_NODES * 64, T), T, 0, stream>>>(out1, asum1, b1, h1bf);

    // ---- layer 2 ----
    k_gemm2<<<cdiv((long)M2_TILES * 32, T), T, 0, stream>>>(h1bf, w2t, feat2);
    k_logits2<<<cdiv(N_NODES * 4, T), T, 0, stream>>>(feat2, al2, ar2, el2, er2);
    k_edge_max<4><<<cdiv(N_EDGES, T), T, 0, stream>>>(src, dst, el2, er2, emax2);
    k_edge_exp<4><<<cdiv(N_EDGES, T), T, 0, stream>>>(src, dst, el2, er2, emax2, asum2, a2);
    k_edge_agg2<<<cdiv((long)N_EDGES * 32, T), T, 0, stream>>>(src, dst, a2, feat2, out);
    k_final2<<<cdiv(N_NODES * 32, T), T, 0, stream>>>(out, asum2, b2);
}
